// GreaseArgMultipleChoiceTransform_455266533931
// MI455X (gfx1250) — matrix operand order repaired and checked by reading, not compiled, not run on hardware
//
#include <hip/hip_runtime.h>
#include <hip/hip_bf16.h>
#include <stdint.h>

// ---------------------------------------------------------------------------
// GreaseArgMultipleChoiceTransform for MI455X (gfx1250, wave32, WMMA + async).
//
//  1) build_top_rows : per-graph bitmask of valid node ids -> ascending top-K
//  2) scan_choice    : block-scan compaction of choice_mask -> lm row indices
//  3) cvt_f32_bf16   : dense_w / out_w fp32 -> bf16 (in ws)
//  4) gemm_wmma<GATHER=1,TANH=1> : x = tanh([lm_sel|gnn_sel] @ dense_w^T + b)
//  5) gemm_wmma<GATHER=0,TANH=0> : out = x @ out_w^T + out_b   (fp32 out)
//
// GEMM: 128x128 block tile, 8 waves, K-step 32, 4-deep LDS ring,
//   bf16 tiles staged with GLOBAL_LOAD_ASYNC_TO_LDS_B64 (ASYNCcnt, 2 tiles in
//   flight), fp32 gather staged via regs + v_cvt_pk_bf16_f32 + ds_store,
//   v_wmma_f32_16x16x32_bf16 with fp32 accumulation.  Pipeline unrolled x4 so
//   all buffer/register-set indices are compile-time constants (no movrels).
// ---------------------------------------------------------------------------

typedef __attribute__((ext_vector_type(4)))  __bf16 v4bf;
typedef __attribute__((ext_vector_type(8)))  __bf16 v8bf;
typedef __attribute__((ext_vector_type(16))) __bf16 v16bf;
typedef __attribute__((ext_vector_type(4)))  float  v4f;
typedef __attribute__((ext_vector_type(8)))  float  v8f;

static __device__ __forceinline__ __bf16 f2bf(float f) { return (__bf16)f; }

#define H_DIM   1024
#define GH_DIM  512
#define N_PER   64
#define TOPK    16
#define K1      (H_DIM + GH_DIM)   // 1536
#define K2      H_DIM              // 1024

// Async copy global(bf16 x4, 8B) -> LDS, tracked by ASYNCcnt.
static __device__ __forceinline__ void async_b64(uint32_t lds_off,
                                                 const void* gptr) {
  unsigned long long ga = (unsigned long long)gptr;
  asm volatile("global_load_async_to_lds_b64 %0, %1, off"
               :
               : "v"(lds_off), "v"(ga)
               : "memory");
}

// -------------------------------------------------------------------------
// Kernel 1: per-graph ascending top-K node ids -> global gnn row indices.
// -------------------------------------------------------------------------
__global__ void build_top_rows(const int* __restrict__ top_mask,
                               int* __restrict__ gnn_rows, int G) {
  int g = blockIdx.x * blockDim.x + threadIdx.x;
  if (g >= G) return;
  unsigned long long mask = 0ull;
  #pragma unroll 4
  for (int j = 0; j < N_PER; ++j) {
    int v = top_mask[g * N_PER + j];
    if (v >= 0) mask |= (1ull << (v & 63));
  }
  #pragma unroll
  for (int i = 0; i < TOPK; ++i) {
    int pos = 0;
    if (mask) {
      pos = __ffsll((long long)mask) - 1;
      mask &= (mask - 1);
    }
    gnn_rows[g * TOPK + i] = g * N_PER + pos;
  }
}

// -------------------------------------------------------------------------
// Kernel 2: compact choice_mask -> first M true indices (single workgroup).
// -------------------------------------------------------------------------
__global__ void scan_choice(const unsigned char* __restrict__ mask,
                            int* __restrict__ lm_rows, int T, int M) {
  __shared__ int cnt[1024];
  const int tid   = threadIdx.x;
  const int chunk = T / 1024;
  const int base  = tid * chunk;

  for (int i = tid; i < M; i += 1024) lm_rows[i] = 0;   // poison-proof default

  int c = 0;
  for (int j = 0; j < chunk; ++j) c += (mask[base + j] != 0) ? 1 : 0;
  cnt[tid] = c;
  __syncthreads();
  for (int off = 1; off < 1024; off <<= 1) {
    int v = (tid >= off) ? cnt[tid - off] : 0;
    __syncthreads();
    cnt[tid] += v;
    __syncthreads();
  }
  int w = cnt[tid] - c;
  for (int j = 0; j < chunk; ++j) {
    if (mask[base + j] != 0) {
      if (w < M) lm_rows[w] = base + j;
      ++w;
    }
  }
}

// -------------------------------------------------------------------------
// Kernel 3: fp32 -> bf16 weight conversion (grid-stride)
// -------------------------------------------------------------------------
__global__ void cvt_f32_bf16(const float* __restrict__ src,
                             __bf16* __restrict__ dst, int n) {
  for (int i = blockIdx.x * blockDim.x + threadIdx.x; i < n;
       i += gridDim.x * blockDim.x)
    dst[i] = f2bf(src[i]);
}

// -------------------------------------------------------------------------
// Kernel 4/5: async-pipelined WMMA GEMM. C[M,N] = act(A[M,K] @ W[N,K]^T + b)
//  Each thread owns 4 rows (tid>>3 + 32j) x 4 cols ((tid&7)*4) of the 128x32
//  tile. bf16 tiles: async b64 global->LDS. fp32 gather: b128 load + cvt +
//  ds_store_b64. 4-buffer LDS ring, 2 tiles in flight on ASYNCcnt, one
//  barrier per K-step (write target's last reader is 2 tiles old, which
//  barrier_{t-1} already guarantees has completed).
// -------------------------------------------------------------------------
template <bool GATHER, bool TANH_BF16>
__global__ __launch_bounds__(256) void gemm_wmma(
    const float*  __restrict__ lm_x,     // [T, H]      (GATHER)
    const float*  __restrict__ gnn_x,    // [G*64, GH]  (GATHER)
    const int*    __restrict__ lm_rows,  // [M]         (GATHER)
    const int*    __restrict__ gnn_rows, // [M]         (GATHER)
    const __bf16* __restrict__ A_bf,     // [M, kdim]   (!GATHER)
    const __bf16* __restrict__ W,        // [Ntot, kdim] bf16
    const float*  __restrict__ bias,     // [Ntot]
    __bf16*       __restrict__ out_bf,   // (TANH_BF16)
    float*        __restrict__ out_f,    // (!TANH_BF16)
    int kdim, int Ntot) {

  constexpr int BM = 128, BN = 128, LDK = 40, KSTEP = 32, NBUF = 4;
  constexpr uint32_t BUFB = (uint32_t)BM * LDK * 2;  // 10240 B per buffer
  __shared__ __align__(16) __bf16 As[NBUF][BM][LDK];
  __shared__ __align__(16) __bf16 Bs[NBUF][BN][LDK];

  const int tid  = threadIdx.x;
  const int wave = tid >> 5;
  const int lane = tid & 31;
  const int hi   = lane >> 4;
  const int mloc = lane & 15;

  const int m0 = blockIdx.y * BM;
  const int n0 = blockIdx.x * BN;

  // ---- per-thread staging geometry (4 rows x 4 cols of the 128x32 tile) ----
  const int srow  = tid >> 3;        // 0..31, rows srow + 32j
  const int scol4 = (tid & 7) * 4;   // 0,4,...,28

  uint32_t lmoff[4], gnoff[4];
  const __bf16* aptr[4];
  const __bf16* bptr[4];
  uint32_t alds[4], blds[4];
  #pragma unroll
  for (int j = 0; j < 4; ++j) {
    int r = srow + 32 * j;
    if (GATHER) {
      lmoff[j] = (uint32_t)lm_rows[m0 + r] * H_DIM + scol4;
      gnoff[j] = (uint32_t)gnn_rows[m0 + r] * GH_DIM + scol4;
    } else {
      aptr[j] = A_bf + (uint32_t)(m0 + r) * (uint32_t)kdim + scol4;
      alds[j] = (uint32_t)(uintptr_t)&As[0][r][scol4];
    }
    bptr[j] = W + (uint32_t)(n0 + r) * (uint32_t)kdim + scol4;
    blds[j] = (uint32_t)(uintptr_t)&Bs[0][r][scol4];
  }

  const int wrow = (wave & 3) * 2;   // row subtiles: wrow, wrow+1
  const int wcol = (wave >> 2) * 4;  // col subtiles: wcol..wcol+3

  const v8f zacc = {};
  v8f acc[2][4];
  #pragma unroll
  for (int i = 0; i < 2; ++i)
    #pragma unroll
    for (int j = 0; j < 4; ++j) acc[i][j] = zacc;

  v4f aF[2][4];   // GATHER: fp32 gather staging (ping-pong, static indices)

  // --- async issue of one bf16 tile pair (B always, A when !GATHER) ---
  auto issueAsync = [&](int k0, int buf) {
    uint32_t bo = (uint32_t)buf * BUFB;
    #pragma unroll
    for (int j = 0; j < 4; ++j)
      async_b64(blds[j] + bo, (const void*)(bptr[j] + k0));
    if (!GATHER) {
      #pragma unroll
      for (int j = 0; j < 4; ++j)
        async_b64(alds[j] + bo, (const void*)(aptr[j] + k0));
    }
  };
  // wait until at most `pend` tiles of async copies remain in flight
  auto waitAsync = [&](int pend) {
    if (GATHER) {            // 4 async instructions per wave per tile
      if (pend >= 2)      asm volatile("s_wait_asynccnt 0x8" ::: "memory");
      else if (pend == 1) asm volatile("s_wait_asynccnt 0x4" ::: "memory");
      else                asm volatile("s_wait_asynccnt 0x0" ::: "memory");
    } else {                 // 8 async instructions per wave per tile
      if (pend >= 2)      asm volatile("s_wait_asynccnt 0x10" ::: "memory");
      else if (pend == 1) asm volatile("s_wait_asynccnt 0x8" ::: "memory");
      else                asm volatile("s_wait_asynccnt 0x0" ::: "memory");
    }
  };

  auto loadA = [&](int k0, int s) {   // GATHER only
    if (k0 < H_DIM) {
      #pragma unroll
      for (int j = 0; j < 4; ++j) aF[s][j] = *(const v4f*)(lm_x + lmoff[j] + k0);
    } else {
      #pragma unroll
      for (int j = 0; j < 4; ++j)
        aF[s][j] = *(const v4f*)(gnn_x + gnoff[j] + (k0 - H_DIM));
    }
  };
  auto storeA = [&](int buf, int s) {  // GATHER only
    #pragma unroll
    for (int j = 0; j < 4; ++j) {
      v4bf t;
      #pragma unroll
      for (int e = 0; e < 4; ++e) t[e] = f2bf(aF[s][j][e]);
      *(v4bf*)&As[buf][srow + 32 * j][scol4] = t;
    }
  };

  auto computeTile = [&](int buf) {
    v16bf afrag[2];
    #pragma unroll
    for (int i = 0; i < 2; ++i) {
      int rr = (wrow + i) * 16 + mloc;
      v8bf lo = *(const v8bf*)&As[buf][rr][hi * 8];
      v8bf hp = *(const v8bf*)&As[buf][rr][16 + hi * 8];
      afrag[i] = __builtin_shufflevector(lo, hp, 0, 1, 2, 3, 4, 5, 6, 7, 8, 9,
                                         10, 11, 12, 13, 14, 15);
    }
    v16bf bfrag[4];
    #pragma unroll
    for (int j = 0; j < 4; ++j) {
      int nn = (wcol + j) * 16 + mloc;
      v8bf lo = *(const v8bf*)&Bs[buf][nn][hi * 8];
      v8bf hp = *(const v8bf*)&Bs[buf][nn][16 + hi * 8];
      bfrag[j] = __builtin_shufflevector(lo, hp, 0, 1, 2, 3, 4, 5, 6, 7, 8, 9,
                                         10, 11, 12, 13, 14, 15);
    }
    #pragma unroll
    for (int i = 0; i < 2; ++i)
      #pragma unroll
      for (int j = 0; j < 4; ++j)
        acc[i][j] = __builtin_amdgcn_wmma_f32_16x16x32_bf16(
            false, afrag[i], false, bfrag[j], (short)0, acc[i][j], false,
            false);
  };

  // ---- pipeline over K: 2 async tiles in flight, 4-buffer LDS ring ----
  // Unrolled x4: buffer index (phase) and A-register set (phase&1) are
  // compile-time constants.  nk is 48 or 32 (divisible by 4).
  const int nk = kdim / KSTEP;
  issueAsync(0, 0);
  if (nk > 1) issueAsync(KSTEP, 1);
  if (GATHER) loadA(0, 0);

#define PIPE_STEP(P)                                                        \
  {                                                                         \
    const int tt = t + (P);                                                 \
    if (GATHER) {                                                           \
      storeA((P) & 3, (P) & 1);                                             \
      if (tt + 1 < nk) loadA((tt + 1) * KSTEP, ((P) + 1) & 1);              \
    }                                                                       \
    if (tt + 2 < nk) issueAsync((tt + 2) * KSTEP, ((P) + 2) & 3);           \
    int rem = nk - 1 - tt;                                                  \
    waitAsync(rem > 2 ? 2 : rem);                                           \
    __syncthreads();                                                        \
    computeTile((P) & 3);                                                   \
  }

  for (int t = 0; t < nk; t += 4) {
    PIPE_STEP(0)
    PIPE_STEP(1)
    PIPE_STEP(2)
    PIPE_STEP(3)
  }
#undef PIPE_STEP

  // ---- epilogue: bias (+tanh -> bf16) or plain fp32 store ----
  #pragma unroll
  for (int i = 0; i < 2; ++i) {
    #pragma unroll
    for (int j = 0; j < 4; ++j) {
      #pragma unroll
      for (int r = 0; r < 8; ++r) {
        int row = m0 + (wrow + i) * 16 + hi * 8 + r;
        int col = n0 + (wcol + j) * 16 + mloc;
        float v = acc[i][j][r] + bias[col];
        if (TANH_BF16)
          out_bf[(size_t)row * Ntot + col] = f2bf(tanhf(v));
        else
          out_f[(size_t)row * Ntot + col] = v;
      }
    }
  }
}

// ---------------------------------------------------------------------------
// Host-side launcher.  Inputs (setup_inputs order):
//  0 lm_x[T,H] f32   1 choice_mask[T] bool   2 gnn_x[G*64,GH] f32
//  3 top_mask[G*64] i32   4 ptr[G+1] i32   5 k i32
//  6 dense_w[H,H+GH] f32  7 dense_b[H] f32  8 out_w[H,H] f32  9 out_b[H] f32
// ---------------------------------------------------------------------------
extern "C" void kernel_launch(void* const* d_in, const int* in_sizes, int n_in,
                              void* d_out, int out_size, void* d_ws,
                              size_t ws_size, hipStream_t stream) {
  const float*         lm_x   = (const float*)d_in[0];
  const unsigned char* choice = (const unsigned char*)d_in[1];
  const float*         gnn_x  = (const float*)d_in[2];
  const int*           top_m  = (const int*)d_in[3];
  const float*         dw     = (const float*)d_in[6];
  const float*         db     = (const float*)d_in[7];
  const float*         ow     = (const float*)d_in[8];
  const float*         ob     = (const float*)d_in[9];
  float*               out    = (float*)d_out;

  const int G = in_sizes[3] / N_PER;     // 2048
  const int T = in_sizes[1];             // 65536
  const int M = G * TOPK;                // 32768

  // workspace layout (256B-aligned slabs), ~70 MB total
  char*   ws       = (char*)d_ws;
  size_t  off      = 0;
  int*    lm_rows  = (int*)(ws + off);           off += (size_t)M * 4;
  int*    gnn_rows = (int*)(ws + off);           off += (size_t)M * 4;
  off = (off + 255) & ~(size_t)255;
  __bf16* dw_b     = (__bf16*)(ws + off);        off += (size_t)H_DIM * K1 * 2;
  off = (off + 255) & ~(size_t)255;
  __bf16* ow_b     = (__bf16*)(ws + off);        off += (size_t)H_DIM * K2 * 2;
  off = (off + 255) & ~(size_t)255;
  __bf16* xact     = (__bf16*)(ws + off);        off += (size_t)M * H_DIM * 2;
  (void)ws_size;

  build_top_rows<<<(G + 255) / 256, 256, 0, stream>>>(top_m, gnn_rows, G);
  scan_choice<<<1, 1024, 0, stream>>>(choice, lm_rows, T, M);

  cvt_f32_bf16<<<512, 256, 0, stream>>>(dw, dw_b, H_DIM * K1);
  cvt_f32_bf16<<<512, 256, 0, stream>>>(ow, ow_b, H_DIM * K2);

  {
    dim3 grid(H_DIM / 128, M / 128);   // (8, 256)
    gemm_wmma<true, true><<<grid, 256, 0, stream>>>(
        lm_x, gnn_x, lm_rows, gnn_rows, (const __bf16*)nullptr, dw_b, db, xact,
        (float*)nullptr, K1, H_DIM);
  }
  {
    dim3 grid(H_DIM / 128, M / 128);   // (8, 256)
    gemm_wmma<false, false><<<grid, 256, 0, stream>>>(
        (const float*)nullptr, (const float*)nullptr, (const int*)nullptr,
        (const int*)nullptr, xact, ow_b, ob, (__bf16*)nullptr, out, K2, H_DIM);
  }
  (void)n_in; (void)out_size;
}
